// sLSTMCell_2551210574034
// MI455X (gfx1250) — compile-verified, matrix operand order checked by reading, not run on hardware
//
#include <hip/hip_runtime.h>
#include <hip/hip_bf16.h>

typedef __attribute__((ext_vector_type(8)))  float  v8f;
typedef __attribute__((ext_vector_type(8)))  __bf16 v8bf;
typedef __attribute__((ext_vector_type(16))) __bf16 v16bf;

#define B_DIM 4096
#define I_DIM 1024
#define H_DIM 2048
#define H4    8192
#define KSTEP 32
#define STEPS ((I_DIM + H_DIM) / KSTEP)   // 32 + 64 = 96
#define LDT   (128 * 48)                  // one LDS tile: 128 rows x 32 k, stride 48

__device__ __forceinline__ unsigned short f2bf_bits(float f) {
  unsigned u = __float_as_uint(f);
  unsigned r = u + 0x7FFFu + ((u >> 16) & 1u);   // round to nearest even
  return (unsigned short)(r >> 16);
}

union Frag { v16bf v; v8bf h[2]; };

// ---------------------------------------------------------------------------
// Kernel 1: pre[B,4H] = x @ Wcat^T + bcat + h_prev @ Rcat^T
// bf16 WMMA, f32 accumulate. 128x128 tile per block, BK=32, 8 waves (2M x 4N),
// each wave 4x2 16x16 WMMA tiles. Double-buffered LDS, 1 barrier per K-step.
// ---------------------------------------------------------------------------
__global__ __launch_bounds__(256)
void slstm_gemm_pre(const float* __restrict__ x, const float* __restrict__ h_prev,
                    const float* __restrict__ Wi, const float* __restrict__ Wf,
                    const float* __restrict__ Wz, const float* __restrict__ Wo,
                    const float* __restrict__ Ri, const float* __restrict__ Rf,
                    const float* __restrict__ Rz, const float* __restrict__ Ro,
                    const float* __restrict__ bi, const float* __restrict__ bfv,
                    const float* __restrict__ bz, const float* __restrict__ bo,
                    float* __restrict__ pre)
{
  __shared__ __bf16 As[2 * LDT];   // 2 x 12 KB
  __shared__ __bf16 Bs[2 * LDT];   // 2 x 12 KB

  const int tileN = blockIdx.x * 128;
  const int tileM = blockIdx.y * 128;
  const int tid   = threadIdx.x;
  const int wave  = tid >> 5;
  const int lane  = tid & 31;
  const int waveM = wave & 1;        // 2 waves along M (64 rows each)
  const int waveN = wave >> 1;       // 4 waves along N (32 cols each)
  const int half  = lane >> 4;
  const int l16   = lane & 15;
  const int klo   = half * 8;        // 16-bit A/B fragment K-chunk select

  // 2048 cols per gate; a 128-wide tile never straddles gates
  const int gate = tileN >> 11;
  const int nrow = tileN & (H_DIM - 1);
  const float* Wg = (gate == 0) ? Wi : (gate == 1) ? Wf : (gate == 2) ? Wz : Wo;
  const float* Rg = (gate == 0) ? Ri : (gate == 1) ? Rf : (gate == 2) ? Rz : Ro;
  const float* bg = (gate == 0) ? bi : (gate == 1) ? bfv : (gate == 2) ? bz : bo;

  // per-thread staging geometry (fixed across steps)
  const int ldr0 = (tid * 4) >> 3;            // not used; keep simple below
  (void)ldr0;

  v8f acc[4][2];
  #pragma unroll
  for (int mi = 0; mi < 4; ++mi)
    #pragma unroll
    for (int ni = 0; ni < 2; ++ni)
      #pragma unroll
      for (int e = 0; e < 8; ++e) acc[mi][ni][e] = 0.f;

  float4 va[4], vb[4];

  // ---- prologue: fetch step 0 and stage into buffer 0 ----
  #pragma unroll
  for (int it = 0; it < 4; ++it) {
    int c  = tid + it * 256;
    int r  = c >> 3;
    int kc = (c & 7) * 4;
    va[it] = *(const float4*)(x  + (size_t)(tileM + r) * I_DIM + kc);
    vb[it] = *(const float4*)(Wg + (size_t)(nrow  + r) * I_DIM + kc);
  }
  #pragma unroll
  for (int it = 0; it < 4; ++it) {
    int c  = tid + it * 256;
    int r  = c >> 3;
    int kc = (c & 7) * 4;
    ushort4 pa, pb;
    pa.x = f2bf_bits(va[it].x); pa.y = f2bf_bits(va[it].y);
    pa.z = f2bf_bits(va[it].z); pa.w = f2bf_bits(va[it].w);
    pb.x = f2bf_bits(vb[it].x); pb.y = f2bf_bits(vb[it].y);
    pb.z = f2bf_bits(vb[it].z); pb.w = f2bf_bits(vb[it].w);
    *(ushort4*)(As + r * 48 + kc) = pa;
    *(ushort4*)(Bs + r * 48 + kc) = pb;
  }
  __syncthreads();

  int cur = 0;
  for (int s = 0; s < STEPS; ++s) {
    // 1) global prefetch for step s+1 (overlaps WMMA below)
    const bool more = (s + 1 < STEPS);
    if (more) {
      int sp = s + 1;
      const float* Ag; const float* Bg; int K, k0;
      if (sp < I_DIM / KSTEP) { Ag = x;      Bg = Wg; K = I_DIM; k0 = sp * KSTEP; }
      else                    { Ag = h_prev; Bg = Rg; K = H_DIM; k0 = (sp - I_DIM / KSTEP) * KSTEP; }
      #pragma unroll
      for (int it = 0; it < 4; ++it) {
        int c  = tid + it * 256;
        int r  = c >> 3;
        int kc = (c & 7) * 4;
        va[it] = *(const float4*)(Ag + (size_t)(tileM + r) * K + k0 + kc);
        vb[it] = *(const float4*)(Bg + (size_t)(nrow  + r) * K + k0 + kc);
      }
    }

    // 2) compute from current LDS buffer
    {
      const __bf16* Asc = As + cur * LDT;
      const __bf16* Bsc = Bs + cur * LDT;
      Frag a[4], b[2];
      #pragma unroll
      for (int mi = 0; mi < 4; ++mi) {
        int row = waveM * 64 + mi * 16 + l16;
        a[mi].h[0] = *(const v8bf*)(Asc + row * 48 + klo);
        a[mi].h[1] = *(const v8bf*)(Asc + row * 48 + 16 + klo);
      }
      #pragma unroll
      for (int ni = 0; ni < 2; ++ni) {
        int col = waveN * 32 + ni * 16 + l16;
        b[ni].h[0] = *(const v8bf*)(Bsc + col * 48 + klo);
        b[ni].h[1] = *(const v8bf*)(Bsc + col * 48 + 16 + klo);
      }
      #pragma unroll
      for (int mi = 0; mi < 4; ++mi)
        #pragma unroll
        for (int ni = 0; ni < 2; ++ni)
          acc[mi][ni] = __builtin_amdgcn_wmma_f32_16x16x32_bf16(
              false, a[mi].v, false, b[ni].v, (short)0, acc[mi][ni], false, false);
    }

    // 3) stage prefetched data into the alternate buffer
    if (more) {
      __bf16* Asn = As + (cur ^ 1) * LDT;
      __bf16* Bsn = Bs + (cur ^ 1) * LDT;
      #pragma unroll
      for (int it = 0; it < 4; ++it) {
        int c  = tid + it * 256;
        int r  = c >> 3;
        int kc = (c & 7) * 4;
        ushort4 pa, pb;
        pa.x = f2bf_bits(va[it].x); pa.y = f2bf_bits(va[it].y);
        pa.z = f2bf_bits(va[it].z); pa.w = f2bf_bits(va[it].w);
        pb.x = f2bf_bits(vb[it].x); pb.y = f2bf_bits(vb[it].y);
        pb.z = f2bf_bits(vb[it].z); pb.w = f2bf_bits(vb[it].w);
        *(ushort4*)(Asn + r * 48 + kc) = pa;
        *(ushort4*)(Bsn + r * 48 + kc) = pb;
      }
    }
    __syncthreads();
    cur ^= 1;
  }

  // epilogue: add bias, store pre tile. C layout: lane holds N=l16, rows half*8+e
  #pragma unroll
  for (int ni = 0; ni < 2; ++ni) {
    int col = tileN + waveN * 32 + ni * 16 + l16;
    float bv = bg[col & (H_DIM - 1)];
    #pragma unroll
    for (int mi = 0; mi < 4; ++mi) {
      int rbase = tileM + waveM * 64 + mi * 16 + half * 8;
      #pragma unroll
      for (int e = 0; e < 8; ++e)
        pre[(size_t)(rbase + e) * H4 + col] = acc[mi][ni][e] + bv;
    }
  }
}

// ---------------------------------------------------------------------------
// Kernel 2: gates + state update + GroupNorm(1) + outputs (h,c,n,m)
// One block per batch row; 256 threads x 8 columns.
// ---------------------------------------------------------------------------
__global__ __launch_bounds__(256)
void slstm_gates_gn(const float* __restrict__ pre,
                    const float* __restrict__ c_prev, const float* __restrict__ n_prev,
                    const float* __restrict__ m_prev,
                    const float* __restrict__ gn_w, const float* __restrict__ gn_b,
                    float* __restrict__ out)
{
  const int row = blockIdx.x;
  const size_t rb = (size_t)row * H_DIM;
  const float* p = pre + (size_t)row * H4;
  float* out_h = out;
  float* out_c = out + (size_t)B_DIM * H_DIM;
  float* out_n = out + 2 * (size_t)B_DIM * H_DIM;
  float* out_m = out + 3 * (size_t)B_DIM * H_DIM;

  float cs[8], ov[8];
  float s = 0.f, ss = 0.f;
  #pragma unroll
  for (int j = 0; j < 8; ++j) {
    int hc = threadIdx.x + j * 256;
    float pi = p[hc];
    float pf = p[H_DIM + hc];
    float pz = p[2 * H_DIM + hc];
    float po = p[3 * H_DIM + hc];
    float ig = __frcp_rn(1.f + __expf(-pi));
    float fg = __frcp_rn(1.f + __expf(-pf));
    float z  = tanhf(pz);
    float og = __frcp_rn(1.f + __expf(-po));
    float mv = fmaxf(fg * m_prev[rb + hc], ig);
    float cv = fg * c_prev[rb + hc] + ig * z;
    float nv = fg * n_prev[rb + hc] + ig;
    float csv = cv / nv;
    out_c[rb + hc] = cv;
    out_n[rb + hc] = nv;
    out_m[rb + hc] = mv;
    cs[j] = csv; ov[j] = og;
    s += csv; ss += csv * csv;
  }

  // wave32 shuffle reduction, then cross-wave via LDS
  #pragma unroll
  for (int off = 16; off > 0; off >>= 1) {
    s  += __shfl_down(s,  off, 32);
    ss += __shfl_down(ss, off, 32);
  }
  __shared__ float red0[8], red1[8];
  int wid = threadIdx.x >> 5, lane = threadIdx.x & 31;
  if (lane == 0) { red0[wid] = s; red1[wid] = ss; }
  __syncthreads();
  if (threadIdx.x == 0) {
    float S = 0.f, SS = 0.f;
    for (int w = 0; w < 8; ++w) { S += red0[w]; SS += red1[w]; }
    float mu  = S * (1.f / H_DIM);
    float var = SS * (1.f / H_DIM) - mu * mu;
    red0[0] = mu;
    red1[0] = __frsqrt_rn(var + 1e-5f);
  }
  __syncthreads();
  float mu = red0[0], inv = red1[0];
  #pragma unroll
  for (int j = 0; j < 8; ++j) {
    int hc = threadIdx.x + j * 256;
    float g = (cs[j] - mu) * inv * gn_w[hc] + gn_b[hc];
    out_h[rb + hc] = ov[j] * tanhf(g);
  }
}

extern "C" void kernel_launch(void* const* d_in, const int* in_sizes, int n_in,
                              void* d_out, int out_size, void* d_ws, size_t ws_size,
                              hipStream_t stream) {
  (void)in_sizes; (void)n_in; (void)out_size; (void)ws_size;
  const float* x      = (const float*)d_in[0];
  const float* h_prev = (const float*)d_in[1];
  const float* c_prev = (const float*)d_in[2];
  const float* n_prev = (const float*)d_in[3];
  const float* m_prev = (const float*)d_in[4];
  const float* Wi = (const float*)d_in[5];
  const float* Wf = (const float*)d_in[6];
  const float* Wz = (const float*)d_in[7];
  const float* Wo = (const float*)d_in[8];
  const float* bi = (const float*)d_in[9];
  const float* bf = (const float*)d_in[10];
  const float* bz = (const float*)d_in[11];
  const float* bo = (const float*)d_in[12];
  const float* Ri = (const float*)d_in[13];
  const float* Rf = (const float*)d_in[14];
  const float* Rz = (const float*)d_in[15];
  const float* Ro = (const float*)d_in[16];
  const float* gn_w = (const float*)d_in[17];
  const float* gn_b = (const float*)d_in[18];

  float* pre = (float*)d_ws;  // B * 4H floats = 128 MiB scratch

  dim3 gemm_grid(H4 / 128, B_DIM / 128);  // 64 x 32 blocks
  slstm_gemm_pre<<<gemm_grid, 256, 0, stream>>>(
      x, h_prev, Wi, Wf, Wz, Wo, Ri, Rf, Rz, Ro, bi, bf, bz, bo, pre);

  slstm_gates_gn<<<B_DIM, 256, 0, stream>>>(
      pre, c_prev, n_prev, m_prev, gn_w, gn_b, (float*)d_out);
}
